// DQN_19198503813318
// MI455X (gfx1250) — matrix operand order repaired and checked by reading, not compiled, not run on hardware
//
#include <hip/hip_runtime.h>

typedef __attribute__((ext_vector_type(16))) _Float16 v16h;
typedef __attribute__((ext_vector_type(8)))  _Float16 v8h;
typedef __attribute__((ext_vector_type(8)))  float    v8f;
typedef _Float16 half_t;

#define BDIM 128
static constexpr int BL = 512 * 6;   // batch * landmarks = 3072

__device__ __forceinline__ float prelu_f(float x, float a) {
    return x >= 0.0f ? x : a * x;
}

__device__ __forceinline__ v8f wmma_f16(v16h a, v16h b, v8f c) {
    return __builtin_amdgcn_wmma_f32_16x16x32_f16(false, a, false, b,
                                                  (short)0, c, false, false);
}

// ---------------------------------------------------------------------------
// Weight pre-swizzle: conv weights (COUT,CIN,KH,KW) f32 -> fragment-order f16
// layout: [kstep][ntile][lane][16 halfs]; k = (ky*KW+kx)*CIN + ci
// B-fragment lane mapping: col n = 16*t + (lane&15); k = 32*s + 8*(lane>>4) + e + (e>=8 ? 8 : 0)
// ---------------------------------------------------------------------------
template<int COUT, int CIN, int KH, int KW, int KPAD>
__global__ __launch_bounds__(256) void prep_conv_w(const float* __restrict__ w,
                                                   half_t* __restrict__ wf) {
    constexpr int NT = COUT / 16;
    constexpr int KREAL = CIN * KH * KW;
    int idx = blockIdx.x * 256 + threadIdx.x;
    if (idx >= KPAD * COUT) return;
    int e    = idx & 15;
    int lane = (idx >> 4) & 31;
    int rest = idx >> 9;
    int t = rest % NT;
    int s = rest / NT;
    int kk = 8 * (lane >> 4) + e + ((e >= 8) ? 8 : 0);
    int gk = 32 * s + kk;
    int n  = 16 * t + (lane & 15);
    float v = 0.0f;
    if (gk < KREAL) {
        int ci  = gk % CIN;
        int pos = gk / CIN;
        int ky = pos / KW, kx = pos % KW;
        v = w[((n * CIN + ci) * KH + ky) * KW + kx];
    }
    wf[idx] = (half_t)v;
}

// fc weights (6,K,N) f32 -> per-l fragment-order f16, N padded to NPAD
template<int K, int N, int NPAD>
__global__ __launch_bounds__(256) void prep_fc_w(const float* __restrict__ w,
                                                 half_t* __restrict__ wf) {
    constexpr int NT = NPAD / 16;
    constexpr int NS = K / 32;
    int idx = blockIdx.x * 256 + threadIdx.x;
    if (idx >= 6 * K * NPAD) return;
    int e    = idx & 15;
    int lane = (idx >> 4) & 31;
    int rest = idx >> 9;
    int t = rest % NT; rest /= NT;
    int s = rest % NS;
    int l = rest / NS;
    int kk = 8 * (lane >> 4) + e + ((e >= 8) ? 8 : 0);
    int gk = 32 * s + kk;
    int n  = 16 * t + (lane & 15);
    float v = (n < N) ? w[((size_t)l * K + gk) * N + n] : 0.0f;
    wf[idx] = (half_t)v;
}

// ---------------------------------------------------------------------------
// Fused implicit-GEMM conv + bias + PReLU + 2x2 maxpool.
// One workgroup = one (image, pooled-row): computes conv rows {2r, 2r+1}
// as an M = 2*OUT_W GEMM on WMMA, C tiles -> LDS, then pools to f16 output.
// Activations (f16) live in HWC layout: [bl][y][x][c].
// ---------------------------------------------------------------------------
template<int CIN, int COUT, int KH, int KW, int PAD, int IN_H, int IN_W,
         int OUT_W, int POUT_H, int POUT_W, int KSTEPS, bool CIN32, bool IN_F32>
__global__ __launch_bounds__(BDIM) void conv_pool_kernel(
        const void* __restrict__ in_raw, const v16h* __restrict__ wf,
        const float* __restrict__ bias, const float* __restrict__ alpha,
        half_t* __restrict__ out) {
    constexpr int ROWS    = KH + 1;
    constexpr int RS      = IN_W * CIN;           // staged row size (halfs)
    constexpr int M_VALID = 2 * OUT_W;
    constexpr int MT      = (M_VALID + 15) / 16;
    constexpr int NT      = COUT / 16;
    constexpr int PAIRS   = MT * NT;
    constexpr int KREAL   = CIN * KH * KW;

    __shared__ half_t s_in[ROWS * RS];            // [row][x][ci]
    __shared__ float  s_out[2 * OUT_W * COUT];    // [row&1][x][co]

    const int bl  = blockIdx.x;
    const int r   = blockIdx.y;
    const int tid = threadIdx.x;
    const int row0 = 2 * r - PAD;

    if constexpr (IN_F32) {
        // conv0: input x is f32 NCHW; per-element convert+transpose to [row][x][ci]
        for (int i = tid; i < ROWS * RS; i += BDIM) {
            int ix  = i % IN_W;
            int rem = i / IN_W;
            int lr  = rem % ROWS;
            int ci  = rem / ROWS;
            int giy = row0 + lr;
            float v = 0.0f;
            if (giy >= 0 && giy < IN_H) {
                size_t src = ((size_t)(bl * CIN + ci) * IN_H + giy) * IN_W + ix;
                v = ((const float*)in_raw)[src];
            }
            s_in[(lr * IN_W + ix) * CIN + ci] = (half_t)v;
        }
    } else {
        // f16 HWC input: contiguous 16B vector copies, zero-fill padded rows
        const half_t* inp = (const half_t*)in_raw;
        for (int i = tid; i < ROWS * RS / 8; i += BDIM) {
            int i8  = i * 8;
            int lr  = i8 / RS;
            int rem = i8 % RS;
            int giy = row0 + lr;
            v8h v = {};
            if (giy >= 0 && giy < IN_H)
                v = *(const v8h*)&inp[((size_t)bl * IN_H + giy) * RS + rem];
            *(v8h*)&s_in[i8] = v;
        }
    }
    __syncthreads();

    const int wave = tid >> 5;
    const int lane = tid & 31;
    const int lh   = lane >> 4;
    const int ln   = lane & 15;
    const float av = alpha[0];

    for (int pair = wave; pair < PAIRS; pair += 4) {
        const int mt = pair / NT;
        const int t  = pair % NT;
        v8f acc = {};
        const int  m   = 16 * mt + ln;    // this lane's A-matrix row
        const int  oyl = m / OUT_W;
        const int  ox  = m % OUT_W;
        const bool mok = (m < M_VALID);

        // software-pipelined B fragments (double buffer across k-steps)
        v16h bcur = wf[(0 * NT + t) * 32 + lane];
        for (int s = 0; s < KSTEPS; ++s) {
            const int sn = (s + 1 < KSTEPS) ? s + 1 : s;
            v16h bnext = wf[(sn * NT + t) * 32 + lane];

            union { v16h v; v8h h[2]; half_t e[16]; } af;
            if constexpr (CIN32) {
                // one kernel tap per 32-wide k-step: contiguous ds_read_b128 x2
                const int ky = s / KW, kx = s % KW;
                const int ix = ox + kx - PAD;
                if (mok && ix >= 0 && ix < IN_W) {
                    const half_t* p = &s_in[((oyl + ky) * IN_W + ix) * CIN];
                    af.h[0] = *(const v8h*)(p + 8 * lh);
                    af.h[1] = *(const v8h*)(p + 16 + 8 * lh);
                } else {
                    v8h z = {};
                    af.h[0] = z; af.h[1] = z;
                }
            } else {
                #pragma unroll
                for (int e = 0; e < 16; ++e) {
                    int kk = 8 * lh + e + ((e >= 8) ? 8 : 0);
                    int gk = 32 * s + kk;
                    float v = 0.0f;
                    if (mok && gk < KREAL) {
                        int ci  = gk % CIN;
                        int pos = gk / CIN;
                        int ky = pos / KW, kx = pos % KW;
                        int ix = ox + kx - PAD;
                        if (ix >= 0 && ix < IN_W)
                            v = (float)s_in[((oyl + ky) * IN_W + ix) * CIN + ci];
                    }
                    af.e[e] = (half_t)v;
                }
            }
            acc = wmma_f16(af.v, bcur, acc);
            bcur = bnext;
        }

        // C tile -> LDS with bias + PReLU (C layout: M = rr + 8*lh, N = ln+16t)
        const int n = 16 * t + ln;
        const float bs = bias[n];
        #pragma unroll
        for (int rr = 0; rr < 8; ++rr) {
            int mm = 16 * mt + rr + 8 * lh;
            if (mm < M_VALID) {
                int oy = mm / OUT_W, oxx = mm % OUT_W;
                s_out[(oy * OUT_W + oxx) * COUT + n] = prelu_f(acc[rr] + bs, av);
            }
        }
    }
    __syncthreads();

    // 2x2 max pool over the computed row-pair, emit f16 HWC (channel-coalesced)
    for (int i = tid; i < POUT_W * COUT; i += BDIM) {
        int co = i % COUT;
        int px = i / COUT;
        float v00 = s_out[(0 * OUT_W + 2 * px    ) * COUT + co];
        float v01 = s_out[(0 * OUT_W + 2 * px + 1) * COUT + co];
        float v10 = s_out[(1 * OUT_W + 2 * px    ) * COUT + co];
        float v11 = s_out[(1 * OUT_W + 2 * px + 1) * COUT + co];
        float mx = fmaxf(fmaxf(v00, v01), fmaxf(v10, v11));
        out[((size_t)(bl * POUT_H + r) * POUT_W + px) * COUT + co] = (half_t)mx;
    }
}

// ---------------------------------------------------------------------------
// conv3 (64->64, 3x3, no pad, 4x4 -> 2x2) as 4 GEMMs: M = image, K = 576, N = 64.
// Input is HWC [bl][sp(16)][ci(64)]. Two k-steps share one spatial tap, so the
// loop runs over taps with ping-pong double buffering of both A and B operands.
// ---------------------------------------------------------------------------
__global__ __launch_bounds__(BDIM) void conv3_kernel(
        const half_t* __restrict__ in,   // [BL][16][64] HWC
        const v16h* __restrict__ wf,     // [18][4][32] fragments
        const float* __restrict__ bias, const float* __restrict__ alpha,
        half_t* __restrict__ out) {      // [BL][256]
    const int p  = blockIdx.y;
    const int py = p >> 1, px = p & 1;
    const int tid  = threadIdx.x;
    const int wave = tid >> 5, lane = tid & 31;
    const int lh = lane >> 4, ln = lane & 15;
    const int blb  = blockIdx.x * 64 + wave * 16;
    const int bl_a = blb + ln;
    const half_t* arow = in + (size_t)bl_a * 1024;   // this lane's image

    auto tap_sp = [&](int pos) {
        int ky = pos / 3, kx = pos % 3;
        return (py + ky) * 4 + (px + kx);            // always in bounds (no pad)
    };

    union AF { v16h v; v8h h[2]; };
    v8f acc[4] = {};
    AF afA, afB;
    v16h bufA[4], bufB[4];

    // prologue: tap 0, k-step 0 (ci 0..31)
    {
        const half_t* bp = arow + tap_sp(0) * 64;
        afA.h[0] = *(const v8h*)(bp + 8 * lh);
        afA.h[1] = *(const v8h*)(bp + 16 + 8 * lh);
        #pragma unroll
        for (int t = 0; t < 4; ++t) bufA[t] = wf[t * 32 + lane];
    }

    for (int pos = 0; pos < 9; ++pos) {
        const int s0 = 2 * pos;
        const half_t* bp = arow + tap_sp(pos) * 64;
        // issue loads for k-step s0+1 (ci 32..63, same tap)
        afB.h[0] = *(const v8h*)(bp + 32 + 8 * lh);
        afB.h[1] = *(const v8h*)(bp + 48 + 8 * lh);
        #pragma unroll
        for (int t = 0; t < 4; ++t) bufB[t] = wf[((s0 + 1) * 4 + t) * 32 + lane];
        if (pos + 1 < 9)
            __builtin_prefetch((const void*)&wf[(size_t)(s0 + 2) * 4 * 32], 0, 1);

        #pragma unroll
        for (int t = 0; t < 4; ++t) acc[t] = wmma_f16(afA.v, bufA[t], acc[t]);

        if (pos + 1 < 9) {   // issue loads for next tap's first k-step
            const half_t* bpn = arow + tap_sp(pos + 1) * 64;
            afA.h[0] = *(const v8h*)(bpn + 8 * lh);
            afA.h[1] = *(const v8h*)(bpn + 16 + 8 * lh);
            #pragma unroll
            for (int t = 0; t < 4; ++t)
                bufA[t] = wf[((s0 + 2) * 4 + t) * 32 + lane];
        }

        #pragma unroll
        for (int t = 0; t < 4; ++t) acc[t] = wmma_f16(afB.v, bufB[t], acc[t]);
    }

    const float av = alpha[0];
    #pragma unroll
    for (int t = 0; t < 4; ++t) {
        const int co = 16 * t + ln;
        const float bs = bias[co];
        #pragma unroll
        for (int rr = 0; rr < 8; ++rr) {
            int blo = blb + rr + 8 * lh;
            out[(size_t)blo * 256 + co * 4 + p] =
                (half_t)prelu_f(acc[t][rr] + bs, av);
        }
    }
}

// ---------------------------------------------------------------------------
// Per-landmark FC head: batched GEMM, M = 512 batch rows per l, K/N templated.
// NT<=4 variants double-buffer the B fragments across k-step pairs.
// ---------------------------------------------------------------------------
template<int K, int N, int NPAD, bool PRELU, bool OUT16>
__global__ __launch_bounds__(BDIM) void fc_kernel(
        const half_t* __restrict__ in,   // [B*6][K] f16
        const v16h* __restrict__ wf,     // [6][K/32][NPAD/16][32] fragments
        const float* __restrict__ bias,  // [6][N]
        const float* __restrict__ alpha, // [6]
        void* __restrict__ outp) {
    constexpr int NS = K / 32;
    constexpr int NT = NPAD / 16;
    __shared__ half_t s_a[64 * K];
    const int l   = blockIdx.y;
    const int b0  = blockIdx.x * 64;
    const int tid = threadIdx.x;

    for (int i = tid; i < 64 * K / 8; i += BDIM) {
        int i8 = i * 8;
        int rr = i8 / K;
        int k  = i8 % K;
        const v8h* src = (const v8h*)&in[((size_t)(b0 + rr) * 6 + l) * K + k];
        *(v8h*)&s_a[i8] = *src;
    }
    __syncthreads();

    const int wave = tid >> 5, lane = tid & 31;
    const int lh = lane >> 4, ln = lane & 15;
    const v16h* wl = wf + (size_t)l * NS * NT * 32;
    v8f acc[NT] = {};
    const int rowl = wave * 16 + ln;
    const half_t* arow = &s_a[rowl * K];

    union AF { v16h v; v8h h[2]; };

    if constexpr (NT <= 4) {
        // ping-pong B (and A) double buffering across k-step pairs (NS is even)
        v16h bufA[NT], bufB[NT];
        AF afA, afB;
        afA.h[0] = *(const v8h*)(arow + 8 * lh);
        afA.h[1] = *(const v8h*)(arow + 16 + 8 * lh);
        #pragma unroll
        for (int t = 0; t < NT; ++t) bufA[t] = wl[t * 32 + lane];

        for (int s = 0; s < NS; s += 2) {
            const half_t* p1 = arow + 32 * (s + 1);
            afB.h[0] = *(const v8h*)(p1 + 8 * lh);
            afB.h[1] = *(const v8h*)(p1 + 16 + 8 * lh);
            #pragma unroll
            for (int t = 0; t < NT; ++t)
                bufB[t] = wl[((s + 1) * NT + t) * 32 + lane];

            #pragma unroll
            for (int t = 0; t < NT; ++t) acc[t] = wmma_f16(afA.v, bufA[t], acc[t]);

            if (s + 2 < NS) {
                const half_t* p2 = arow + 32 * (s + 2);
                afA.h[0] = *(const v8h*)(p2 + 8 * lh);
                afA.h[1] = *(const v8h*)(p2 + 16 + 8 * lh);
                #pragma unroll
                for (int t = 0; t < NT; ++t)
                    bufA[t] = wl[((s + 2) * NT + t) * 32 + lane];
            }

            #pragma unroll
            for (int t = 0; t < NT; ++t) acc[t] = wmma_f16(afB.v, bufB[t], acc[t]);
        }
    } else {
        for (int s = 0; s < NS; ++s) {
            AF af;
            const half_t* p = arow + 32 * s;
            af.h[0] = *(const v8h*)(p + 8 * lh);
            af.h[1] = *(const v8h*)(p + 16 + 8 * lh);

            if (s + 1 < NS)
                __builtin_prefetch((const void*)&wl[(size_t)(s + 1) * NT * 32], 0, 1);
            // one clause of B loads, then WMMAs with staggered waits
            v16h bf[NT];
            #pragma unroll
            for (int t = 0; t < NT; ++t)
                bf[t] = wl[(s * NT + t) * 32 + lane];
            #pragma unroll
            for (int t = 0; t < NT; ++t)
                acc[t] = wmma_f16(af.v, bf[t], acc[t]);
        }
    }

    #pragma unroll
    for (int t = 0; t < NT; ++t) {
        const int n = 16 * t + ln;
        if (n < N) {
            const float bs = bias[l * N + n];
            #pragma unroll
            for (int rr = 0; rr < 8; ++rr) {
                int row = b0 + wave * 16 + rr + 8 * lh;
                float v = acc[t][rr] + bs;
                if (PRELU) v = prelu_f(v, alpha[l]);
                if (OUT16)
                    ((half_t*)outp)[((size_t)row * 6 + l) * N + n] = (half_t)v;
                else
                    ((float*)outp)[((size_t)row * 6 + l) * N + n] = v;
            }
        }
    }
}

// ---------------------------------------------------------------------------
extern "C" void kernel_launch(void* const* d_in, const int* in_sizes, int n_in,
                              void* d_out, int out_size, void* d_ws, size_t ws_size,
                              hipStream_t stream) {
    (void)in_sizes; (void)n_in; (void)out_size; (void)ws_size;
    const float* x    = (const float*)d_in[0];
    const float* w0   = (const float*)d_in[1];
    const float* b0   = (const float*)d_in[2];
    const float* w1   = (const float*)d_in[3];
    const float* b1   = (const float*)d_in[4];
    const float* w2   = (const float*)d_in[5];
    const float* b2   = (const float*)d_in[6];
    const float* w3   = (const float*)d_in[7];
    const float* b3   = (const float*)d_in[8];
    const float* a0   = (const float*)d_in[9];
    const float* a1   = (const float*)d_in[10];
    const float* a2   = (const float*)d_in[11];
    const float* a3   = (const float*)d_in[12];
    const float* fc1w = (const float*)d_in[13];
    const float* fc1b = (const float*)d_in[14];
    const float* a4   = (const float*)d_in[15];
    const float* fc2w = (const float*)d_in[16];
    const float* fc2b = (const float*)d_in[17];
    const float* a5   = (const float*)d_in[18];
    const float* fc3w = (const float*)d_in[19];
    const float* fc3b = (const float*)d_in[20];

    half_t* W = (half_t*)d_ws;   // all offsets in f16 elements, 64B-aligned

    const size_t OFF_WF0 = 0;                                   //  128*32
    const size_t OFF_WF1 = OFF_WF0 + 128 * 32;                  //  800*32
    const size_t OFF_WF2 = OFF_WF1 + 800 * 32;                  //  512*64
    const size_t OFF_WF3 = OFF_WF2 + 512 * 64;                  //  576*64
    const size_t OFF_FC1 = OFF_WF3 + 576 * 64;                  // 6*256*128
    const size_t OFF_FC2 = OFF_FC1 + (size_t)6 * 256 * 128;     // 6*128*64
    const size_t OFF_FC3 = OFF_FC2 + (size_t)6 * 128 * 64;      // 6*64*16
    const size_t OFF_B1  = OFF_FC3 + (size_t)6 * 64 * 16;       // pooled0 HWC
    const size_t OFF_B3  = OFF_B1 + (size_t)BL * 32 * 21 * 21;  // pooled1 HWC
    const size_t OFF_B5  = OFF_B3 + (size_t)BL * 32 * 9 * 9;    // pooled2 HWC
    const size_t OFF_B6  = OFF_B5 + (size_t)BL * 64 * 4 * 4;    // flat 256
    const size_t OFF_B7  = OFF_B6 + (size_t)BL * 256;           // fc1 out
    const size_t OFF_B8  = OFF_B7 + (size_t)BL * 128;           // fc2 out

    // Pre-swizzle all weights into WMMA fragment order (f16).
    prep_conv_w<32,  4, 5, 5, 128><<<(128 * 32 + 255) / 256, 256, 0, stream>>>(w0, W + OFF_WF0);
    prep_conv_w<32, 32, 5, 5, 800><<<(800 * 32 + 255) / 256, 256, 0, stream>>>(w1, W + OFF_WF1);
    prep_conv_w<64, 32, 4, 4, 512><<<(512 * 64 + 255) / 256, 256, 0, stream>>>(w2, W + OFF_WF2);
    prep_conv_w<64, 64, 3, 3, 576><<<(576 * 64 + 255) / 256, 256, 0, stream>>>(w3, W + OFF_WF3);
    prep_fc_w<256, 128, 128><<<(6 * 256 * 128 + 255) / 256, 256, 0, stream>>>(fc1w, W + OFF_FC1);
    prep_fc_w<128,  64,  64><<<(6 * 128 *  64 + 255) / 256, 256, 0, stream>>>(fc2w, W + OFF_FC2);
    prep_fc_w< 64,   4,  16><<<(6 *  64 *  16 + 255) / 256, 256, 0, stream>>>(fc3w, W + OFF_FC3);

    // Conv tower (fused conv + bias + PReLU + 2x2 maxpool), all on WMMA.
    conv_pool_kernel<4, 32, 5, 5, 1, 45, 45, 43, 21, 21, 4, false, true>
        <<<dim3(BL, 21), BDIM, 0, stream>>>((const void*)x, (const v16h*)(W + OFF_WF0), b0, a0, W + OFF_B1);
    conv_pool_kernel<32, 32, 5, 5, 1, 21, 21, 19, 9, 9, 25, true, false>
        <<<dim3(BL, 9), BDIM, 0, stream>>>((const void*)(W + OFF_B1), (const v16h*)(W + OFF_WF1), b1, a1, W + OFF_B3);
    conv_pool_kernel<32, 64, 4, 4, 1, 9, 9, 8, 4, 4, 16, true, false>
        <<<dim3(BL, 4), BDIM, 0, stream>>>((const void*)(W + OFF_B3), (const v16h*)(W + OFF_WF2), b2, a2, W + OFF_B5);
    conv3_kernel<<<dim3(BL / 64, 4), BDIM, 0, stream>>>(
        W + OFF_B5, (const v16h*)(W + OFF_WF3), b3, a3, W + OFF_B6);

    // Per-landmark FC heads (batched WMMA GEMMs).
    fc_kernel<256, 128, 128, true, true><<<dim3(8, 6), BDIM, 0, stream>>>(
        W + OFF_B6, (const v16h*)(W + OFF_FC1), fc1b, a4, W + OFF_B7);
    fc_kernel<128, 64, 64, true, true><<<dim3(8, 6), BDIM, 0, stream>>>(
        W + OFF_B7, (const v16h*)(W + OFF_FC2), fc2b, a5, W + OFF_B8);
    fc_kernel<64, 4, 16, false, false><<<dim3(8, 6), BDIM, 0, stream>>>(
        W + OFF_B8, (const v16h*)(W + OFF_FC3), fc3b, a5, d_out);
}
